// MultiHeadGATLayer_65575560675513
// MI455X (gfx1250) — matrix operand order repaired and checked by reading, not compiled, not run on hardware
//
#include <hip/hip_runtime.h>
#include <hip/hip_bf16.h>

typedef __attribute__((ext_vector_type(16))) __bf16 bf16x16;
typedef __attribute__((ext_vector_type(8)))  __bf16 bf16x8;
typedef __attribute__((ext_vector_type(8)))  float  f32x8;
typedef __attribute__((ext_vector_type(4)))  float  f32x4;
typedef __attribute__((ext_vector_type(4)))  int    i32x4;

#define N_NODES   2048
#define BATCH     4
#define INDIM     256
#define HD        256
#define HEADS     4
#define DHEAD     64
#define NROWS     (BATCH * N_NODES)   /* 8192 */
#define ALPHA_LRELU 0.2f
#define NEG_INF_E  (-9000000000000000.0f)

/* ------------------------------------------------------------------ */
/* Kernel 0: W (256x256 f32, row-major [k][c]) -> Wt bf16 [c][k]      */
/* ------------------------------------------------------------------ */
__global__ void __launch_bounds__(256) prep_wt(const float* __restrict__ W,
                                               __bf16* __restrict__ Wt) {
    int idx = blockIdx.x * 256 + threadIdx.x;   // 0..65535
    int c = idx >> 8;
    int k = idx & 255;
    Wt[(size_t)c * INDIM + k] = (__bf16)W[(size_t)k * HD + c];
}

/* ------------------------------------------------------------------ */
/* Kernel 1: q = h @ W via bf16 WMMA.                                 */
/*   grid: 512 blocks (16-row tiles of the 8192 rows), 128 thr = 4 wv */
/*   wave w handles head w (columns 64w..64w+63).                     */
/*   Writes qt bf16 [c][row] (transposed) and e_src/e_dst [h][row].   */
/* ------------------------------------------------------------------ */
__global__ void __launch_bounds__(128) qgen_kernel(
    const float* __restrict__ h, const __bf16* __restrict__ Wt,
    const float* __restrict__ a,
    __bf16* __restrict__ qt, float* __restrict__ esrc, float* __restrict__ edst)
{
    const int lane = threadIdx.x & 31;
    const int wave = threadIdx.x >> 5;          // head
    const int row0 = blockIdx.x * 16;
    const int m    = lane & 15;                 // A-matrix M row
    const int kbA  = (lane < 16) ? 0 : 16;      // A layout K base
    const int kbB  = (lane < 16) ? 0 : 16;      // B layout K base
    const int nloc = lane & 15;                 // B/C column within tile
    const int moff = (lane < 16) ? 0 : 8;       // C row base

    f32x8 acc[4] = {};
    const float* hrow = h + (size_t)(row0 + m) * INDIM;

    for (int kk = 0; kk < INDIM; kk += 32) {
        /* A: 16x32 bf16 tile of h, per-lane contiguous K runs */
        bf16x16 A;
        {
            const float* p0 = hrow + kk + kbA;
            f32x4 a0 = *(const f32x4*)(p0 + 0);
            f32x4 a1 = *(const f32x4*)(p0 + 4);
            f32x4 a2 = *(const f32x4*)(p0 + 8);
            f32x4 a3 = *(const f32x4*)(p0 + 12);
            #pragma unroll
            for (int j = 0; j < 4; ++j) {
                A[j]      = (__bf16)a0[j];
                A[4 + j]  = (__bf16)a1[j];
                A[8 + j]  = (__bf16)a2[j];
                A[12 + j] = (__bf16)a3[j];
            }
        }
        /* B tiles: 32x16 of Wt (contiguous 16 K per lane) + WMMA */
        #pragma unroll
        for (int t = 0; t < 4; ++t) {
            int c = wave * 64 + t * 16 + nloc;
            bf16x16 B = *(const bf16x16*)(Wt + (size_t)c * INDIM + kk + kbB);
            acc[t] = __builtin_amdgcn_wmma_f32_16x16x32_bf16(
                false, A, false, B, (short)0, acc[t], false, false);
        }
    }

    /* Epilogue: store q^T bf16 + fused e_src/e_dst reductions */
    float rs[8], rd[8];
    #pragma unroll
    for (int r = 0; r < 8; ++r) { rs[r] = 0.f; rd[r] = 0.f; }

    #pragma unroll
    for (int t = 0; t < 4; ++t) {
        int d = t * 16 + nloc;                  // 0..63 within head
        int c = wave * 64 + d;                  // absolute column
        float as = a[d];
        float ad = a[64 + d];
        bf16x8 qb;
        #pragma unroll
        for (int r = 0; r < 8; ++r) {
            float v = acc[t][r];
            qb[r] = (__bf16)v;
            rs[r] += v * as;
            rd[r] += v * ad;
        }
        *(bf16x8*)(qt + (size_t)c * NROWS + row0 + moff) = qb;
    }
    /* reduce the 16 columns held across each 16-lane half */
    #pragma unroll
    for (int r = 0; r < 8; ++r) {
        #pragma unroll
        for (int msk = 1; msk <= 8; msk <<= 1) {
            rs[r] += __shfl_xor(rs[r], msk, 32);
            rd[r] += __shfl_xor(rd[r], msk, 32);
        }
    }
    if (nloc == 0) {
        #pragma unroll
        for (int r = 0; r < 8; ++r) {
            int row = row0 + moff + r;
            esrc[wave * NROWS + row] = rs[r];
            edst[wave * NROWS + row] = rd[r];
        }
    }
}

/* ------------------------------------------------------------------ */
/* Kernel 2: masked softmax + attn @ q via bf16 WMMA, ELU fused.      */
/*   grid: 512 blocks = (b, i-tile of 16 rows); 4 waves = 4 heads.    */
/* ------------------------------------------------------------------ */
__global__ void __launch_bounds__(128) attn_kernel(
    const __bf16* __restrict__ qt, const int* __restrict__ adj,
    const float* __restrict__ esrc, const float* __restrict__ edst,
    float* __restrict__ out)
{
    const int lane = threadIdx.x & 31;
    const int head = threadIdx.x >> 5;
    const int b  = blockIdx.x >> 7;             // 0..3
    const int it = blockIdx.x & 127;
    const int i0 = it * 16;
    const int m  = lane & 15;
    const int i  = i0 + m;                      // node row 0..2047

    const float es = esrc[head * NROWS + b * N_NODES + i];
    const int*   adjrow = adj  + (size_t)i * N_NODES;
    const float* edrow  = edst + head * NROWS + b * N_NODES;

    /* ---- pass 1: row max then sum-of-exp (halves split j range) ---- */
    const int jstart = (lane < 16) ? 0 : 1024;
    float mx = -INFINITY;
    for (int j = jstart; j < jstart + 1024; j += 4) {
        /* stream ahead: lowers to global_prefetch_b8 */
        __builtin_prefetch(adjrow + j + 256, 0, 0);
        __builtin_prefetch(edrow  + j + 256, 0, 0);
        i32x4 ad = *(const i32x4*)(adjrow + j);
        f32x4 ed = *(const f32x4*)(edrow + j);
        #pragma unroll
        for (int jj = 0; jj < 4; ++jj) {
            float e = es + ed[jj];
            e = (e >= 0.f) ? e : ALPHA_LRELU * e;
            e = (ad[jj] > 0) ? e : NEG_INF_E;
            mx = fmaxf(mx, e);
        }
    }
    float sum = 0.f;
    for (int j = jstart; j < jstart + 1024; j += 4) {
        i32x4 ad = *(const i32x4*)(adjrow + j);
        f32x4 ed = *(const f32x4*)(edrow + j);
        #pragma unroll
        for (int jj = 0; jj < 4; ++jj) {
            float e = es + ed[jj];
            e = (e >= 0.f) ? e : ALPHA_LRELU * e;
            e = (ad[jj] > 0) ? e : NEG_INF_E;
            sum += __expf(e - mx);
        }
    }
    float mo = __shfl_xor(mx, 16, 32);
    float so = __shfl_xor(sum, 16, 32);
    float M  = fmaxf(mx, mo);
    float S  = sum * __expf(mx - M) + so * __expf(mo - M);
    float rinv = 1.0f / S;

    /* ---- pass 2: P(bf16, normalized) @ q^T tiles via WMMA ---- */
    const int kbA  = (lane < 16) ? 0 : 16;
    const int kbB  = (lane < 16) ? 0 : 16;
    const int nloc = lane & 15;
    f32x8 acc[4] = {};
    const __bf16* qbase = qt + (size_t)(head * 64) * NROWS + b * N_NODES;

    for (int j0 = 0; j0 < N_NODES; j0 += 32) {
        bf16x16 A;
        #pragma unroll
        for (int g = 0; g < 2; ++g) {
            int jb = j0 + kbA + g * 8;
            i32x4 ad0 = *(const i32x4*)(adjrow + jb);
            i32x4 ad1 = *(const i32x4*)(adjrow + jb + 4);
            f32x4 ed0 = *(const f32x4*)(edrow + jb);
            f32x4 ed1 = *(const f32x4*)(edrow + jb + 4);
            #pragma unroll
            for (int jj = 0; jj < 4; ++jj) {
                float e = es + ed0[jj];
                e = (e >= 0.f) ? e : ALPHA_LRELU * e;
                e = (ad0[jj] > 0) ? e : NEG_INF_E;
                A[g * 8 + jj] = (__bf16)(__expf(e - M) * rinv);

                float e2 = es + ed1[jj];
                e2 = (e2 >= 0.f) ? e2 : ALPHA_LRELU * e2;
                e2 = (ad1[jj] > 0) ? e2 : NEG_INF_E;
                A[g * 8 + 4 + jj] = (__bf16)(__expf(e2 - M) * rinv);
            }
        }
        #pragma unroll
        for (int t = 0; t < 4; ++t) {
            const __bf16* bp = qbase + (size_t)(t * 16 + nloc) * NROWS + j0 + kbB;
            bf16x16 B = *(const bf16x16*)bp;
            acc[t] = __builtin_amdgcn_wmma_f32_16x16x32_bf16(
                false, A, false, B, (short)0, acc[t], false, false);
        }
    }

    /* ---- ELU + store (C layout: col = 16t+nloc, rows = moff+r) ---- */
    const int moff = (lane < 16) ? 0 : 8;
    #pragma unroll
    for (int t = 0; t < 4; ++t) {
        int c = head * 64 + t * 16 + nloc;
        #pragma unroll
        for (int r = 0; r < 8; ++r) {
            float x = acc[t][r];
            x = (x > 0.f) ? x : (__expf(x) - 1.0f);
            out[((size_t)b * N_NODES + i0 + moff + r) * HD + c] = x;
        }
    }
}

/* ------------------------------------------------------------------ */
extern "C" void kernel_launch(void* const* d_in, const int* in_sizes, int n_in,
                              void* d_out, int out_size, void* d_ws, size_t ws_size,
                              hipStream_t stream) {
    const float* h   = (const float*)d_in[0];
    const int*   adj = (const int*)d_in[1];
    const float* W   = (const float*)d_in[2];
    const float* a   = (const float*)d_in[3];
    float* out = (float*)d_out;

    char* ws = (char*)d_ws;
    __bf16* qt   = (__bf16*)ws;                                  /* 4 MB   */
    __bf16* Wt   = (__bf16*)(ws + 4u * 1024u * 1024u);           /* 128 KB */
    float*  esrc = (float*)(ws + 4u * 1024u * 1024u + 128u * 1024u);
    float*  edst = (float*)(ws + 4u * 1024u * 1024u + 256u * 1024u);

    hipLaunchKernelGGL(prep_wt,     dim3(256), dim3(256), 0, stream, W, Wt);
    hipLaunchKernelGGL(qgen_kernel, dim3(512), dim3(128), 0, stream, h, Wt, a, qt, esrc, edst);
    hipLaunchKernelGGL(attn_kernel, dim3(512), dim3(128), 0, stream, qt, adj, esrc, edst, out);
}
